// RecurrentNetTimeVariable_4733053960303
// MI455X (gfx1250) — matrix operand order repaired and checked by reading, not compiled, not run on hardware
//
#include <hip/hip_runtime.h>
#include <hip/hip_bf16.h>

typedef __bf16 bf16;
typedef __attribute__((ext_vector_type(16))) __bf16 v16bf;
typedef __attribute__((ext_vector_type(8)))  __bf16 v8bf;
typedef __attribute__((ext_vector_type(8)))  float  v8f;
typedef __attribute__((ext_vector_type(4)))  unsigned int u32x4;
typedef __attribute__((ext_vector_type(8)))  int i32x8;
typedef __attribute__((ext_vector_type(4)))  int i32x4;

#define B_    64
#define T_    512
#define NIN   256
#define NHID  1024
#define NOUT  256

#define LDS_STRIDE 40   // 32 payload + 8 pad bf16 per row (=16+4 DWORDs)

#if __has_builtin(__builtin_amdgcn_tensor_load_to_lds)
#define USE_TDM 1
#else
#define USE_TDM 0
#endif

// ---------- fragment gathers (gfx1250 wave32 WMMA bf16 layouts) ----------

// A-operand 16x32: lanes 0-15 hold K {0..7,16..23}, lanes 16-31 K {8..15,24..31}
__device__ __forceinline__ v16bf frag_a(const bf16* row, int k0, int laneHi) {
  const bf16* p = row + k0 + (laneHi ? 8 : 0);
  v8bf lo = *(const v8bf*)(p);
  v8bf hi = *(const v8bf*)(p + 16);
  return __builtin_shufflevector(lo, hi, 0,1,2,3,4,5,6,7,8,9,10,11,12,13,14,15);
}

// B-operand 32x16: lane n (0-15) holds K {0..7,16..23} of column n, lane n+16 holds K {8..15,24..31}
__device__ __forceinline__ v16bf frag_b(const bf16* wrow, int laneHi) {
  const bf16* p = wrow + (laneHi ? 8 : 0);
  v8bf lo = *(const v8bf*)(p);
  v8bf hi = *(const v8bf*)(p + 16);
  return __builtin_shufflevector(lo, hi, 0,1,2,3,4,5,6,7,8,9,10,11,12,13,14,15);
}

#if USE_TDM
// ---------- TDM: DMA a 64-row x 32-elem bf16 tile (row stride ldK elems) into
// LDS at byte offset lds_off, padding each 16-DWORD row with 4 DWORDs so the
// LDS layout is exactly [64][LDS_STRIDE] bf16.  (D# packing per ISA 8.3/8.4)
__device__ __forceinline__ void tdm_load_tile(unsigned lds_off, const bf16* src,
                                              unsigned ldK) {
  unsigned long long ga = (unsigned long long)(uintptr_t)src;
  u32x4 g0;
  g0[0] = 1u;                                      // count=1, user mode
  g0[1] = lds_off;                                 // lds_addr
  g0[2] = (unsigned)(ga & 0xFFFFFFFFu);            // global_addr[31:0]
  g0[3] = (unsigned)((ga >> 32) & 0x1FFFFFFu)      // global_addr[56:32]
        | (2u << 30);                              // type=2 ("image")
  i32x8 g1;
  // data_size=1 (2B), pad_enable=1, pad_interval=3 (16 DW), pad_amount=3 (4 DW)
  g1[0] = (1 << 16) | (1 << 20) | (3 << 22) | (3 << 25);
  unsigned td0 = ldK;                              // tensor_dim0 (row length)
  unsigned td1 = 1u << 20;                         // tensor_dim1 (rows, large)
  g1[1] = (int)((td0 & 0xFFFFu) << 16);            // barrier addr=0 | td0.lo
  g1[2] = (int)((td0 >> 16) | ((td1 & 0xFFFFu) << 16));
  g1[3] = (int)((td1 >> 16) | (32u << 16));        // tile_dim0 = 32 elems
  g1[4] = (int)(64u);                              // tile_dim1 = 64 rows, dim2=0
  g1[5] = (int)ldK;                                // tensor_dim0_stride lo
  g1[6] = 0;                                       // stride hi | dim1_stride lo
  g1[7] = 0;
  i32x4 z4 = {0, 0, 0, 0};
  i32x8 z8 = {0, 0, 0, 0, 0, 0, 0, 0};
  // 6-arg form (amdgpu-toolchain / clang-23): groups 2,3 unused, last group 0
  __builtin_amdgcn_tensor_load_to_lds(g0, g1, z4, z4, z8, 0);
}
#endif

// ---------- generic C = A[MxK] * W[NxK]^T + bias, bf16 in, f32 acc ----------
// block: 128 threads (4 waves), tile 64(M) x 64(N); grid = (N/64, M/64)

__global__ __launch_bounds__(128)
void gemm_wmma_bf16(const bf16* __restrict__ A, const bf16* __restrict__ W,
                    const float* __restrict__ bias,
                    float* __restrict__ Cf, bf16* __restrict__ Cb,
                    int M, int N, int K) {
  __shared__ bf16 sB[2][64 * LDS_STRIDE];
  const int tid    = threadIdx.x;
  const int lane   = tid & 31;
  const int wave   = tid >> 5;
  const int laneHi = lane >> 4;
  const int lane15 = lane & 15;
  const int n0 = blockIdx.x * 64;
  const int m0 = blockIdx.y * 64;

  v8f zero = {0.f,0.f,0.f,0.f,0.f,0.f,0.f,0.f};
  v8f acc[4] = {zero, zero, zero, zero};

  const bf16* Arow = A + (size_t)(m0 + wave * 16 + lane15) * K;
  const bf16* Wtile = W + (size_t)n0 * K;

#if USE_TDM
  const unsigned lb0 = (unsigned)(uintptr_t)&sB[0][0];
  const unsigned lb1 = (unsigned)(uintptr_t)&sB[1][0];
  if (wave == 0) {                       // prologue: fill buffer 0
    tdm_load_tile(lb0, Wtile, (unsigned)K);
    __builtin_amdgcn_s_wait_tensorcnt(0);
  }
  __syncthreads();
#endif

  for (int k0 = 0; k0 < K; k0 += 32) {
#if USE_TDM
    const int cur = (k0 >> 5) & 1;
    if (k0 + 32 < K && wave == 0)        // async: DMA next tile during compute
      tdm_load_tile(cur ? lb0 : lb1, Wtile + k0 + 32, (unsigned)K);
    const bf16* sT = &sB[cur][0];
#else
    __syncthreads();
    {   // cooperative staging fallback
      int r = tid >> 1;
      int h = (tid & 1) * 16;
      const bf16* src = Wtile + (size_t)r * K + k0 + h;
      bf16* dst = &sB[0][0] + r * LDS_STRIDE + h;
      *(v8bf*)(dst)     = *(const v8bf*)(src);
      *(v8bf*)(dst + 8) = *(const v8bf*)(src + 8);
    }
    __syncthreads();
    const bf16* sT = &sB[0][0];
#endif

    if (k0 + 32 < K) __builtin_prefetch(Arow + k0 + 32, 0, 3);
    v16bf a = frag_a(Arow, k0, laneHi);
#pragma unroll
    for (int nt = 0; nt < 4; ++nt) {
      v16bf b = frag_b(sT + (nt * 16 + lane15) * LDS_STRIDE, laneHi);
      acc[nt] = __builtin_amdgcn_wmma_f32_16x16x32_bf16(
          false, a, false, b, (short)0, acc[nt], false, false);
    }

#if USE_TDM
    __syncthreads();                     // everyone done reading sB[cur]
    if (k0 + 32 < K && wave == 0)
      __builtin_amdgcn_s_wait_tensorcnt(0);
    __syncthreads();                     // next tile resident
#endif
  }

  const int rb = laneHi * 8;
#pragma unroll
  for (int nt = 0; nt < 4; ++nt) {
    int col = n0 + nt * 16 + lane15;
    float bv = bias ? bias[col] : 0.f;
#pragma unroll
    for (int j = 0; j < 8; ++j) {
      int row = m0 + wave * 16 + rb + j;
      float v = acc[nt][j] + bv;
      if (Cf) Cf[(size_t)row * N + col] = v;
      if (Cb) Cb[(size_t)row * N + col] = (bf16)v;
    }
  }
}

// ---------- fused RNN step: h' = (1-a)*h + a*relu(pre_t + h@Whh^T + b_hh) ----
// M = B_ = 64 rows fixed; grid = NHID/64 blocks of 128 threads

__global__ __launch_bounds__(128)
void rnn_step(const bf16* __restrict__ hb_in, const float* __restrict__ hf_in,
              const bf16* __restrict__ Whh, const float* __restrict__ pre,
              const float* __restrict__ b_hh, const float* __restrict__ alpha,
              bf16* __restrict__ hb_out, float* __restrict__ hf_out,
              float* __restrict__ out_hidden, bf16* __restrict__ hlist_bf,
              float* __restrict__ h_last, int t, int is_last) {
  __shared__ bf16 sB[2][64 * LDS_STRIDE];
  const int tid    = threadIdx.x;
  const int lane   = tid & 31;
  const int wave   = tid >> 5;
  const int laneHi = lane >> 4;
  const int lane15 = lane & 15;
  const int n0 = blockIdx.x * 64;

  v8f zero = {0.f,0.f,0.f,0.f,0.f,0.f,0.f,0.f};
  v8f acc[4] = {zero, zero, zero, zero};

  const bf16* Arow  = hb_in + (size_t)(wave * 16 + lane15) * NHID;
  const bf16* Wtile = Whh + (size_t)n0 * NHID;

#if USE_TDM
  const unsigned lb0 = (unsigned)(uintptr_t)&sB[0][0];
  const unsigned lb1 = (unsigned)(uintptr_t)&sB[1][0];
  if (wave == 0) {
    tdm_load_tile(lb0, Wtile, NHID);
    __builtin_amdgcn_s_wait_tensorcnt(0);
  }
  __syncthreads();
#endif

  for (int k0 = 0; k0 < NHID; k0 += 32) {
#if USE_TDM
    const int cur = (k0 >> 5) & 1;
    if (k0 + 32 < NHID && wave == 0)
      tdm_load_tile(cur ? lb0 : lb1, Wtile + k0 + 32, NHID);
    const bf16* sT = &sB[cur][0];
#else
    __syncthreads();
    {
      int r = tid >> 1;
      int h = (tid & 1) * 16;
      const bf16* src = Wtile + (size_t)r * NHID + k0 + h;
      bf16* dst = &sB[0][0] + r * LDS_STRIDE + h;
      *(v8bf*)(dst)     = *(const v8bf*)(src);
      *(v8bf*)(dst + 8) = *(const v8bf*)(src + 8);
    }
    __syncthreads();
    const bf16* sT = &sB[0][0];
#endif

    v16bf a = frag_a(Arow, k0, laneHi);
#pragma unroll
    for (int nt = 0; nt < 4; ++nt) {
      v16bf b = frag_b(sT + (nt * 16 + lane15) * LDS_STRIDE, laneHi);
      acc[nt] = __builtin_amdgcn_wmma_f32_16x16x32_bf16(
          false, a, false, b, (short)0, acc[nt], false, false);
    }

#if USE_TDM
    __syncthreads();
    if (k0 + 32 < NHID && wave == 0)
      __builtin_amdgcn_s_wait_tensorcnt(0);
    __syncthreads();
#endif
  }

  const int rb = laneHi * 8;
#pragma unroll
  for (int nt = 0; nt < 4; ++nt) {
    int col = n0 + nt * 16 + lane15;           // hidden unit g
    float bh = b_hh[col];
    float al = alpha[col];
#pragma unroll
    for (int j = 0; j < 8; ++j) {
      int row = wave * 16 + rb + j;            // batch b
      size_t oidx = ((size_t)row * T_ + t) * NHID + col;
      float cell = acc[nt][j] + pre[oidx] + bh;
      cell = fmaxf(cell, 0.f);
      float hold = hf_in[(size_t)row * NHID + col];
      float hn = hold + al * (cell - hold);
      hf_out[(size_t)row * NHID + col] = hn;
      hb_out[(size_t)row * NHID + col] = (bf16)hn;
      out_hidden[oidx] = hn;
      hlist_bf[oidx]   = (bf16)hn;
      if (is_last) h_last[(size_t)row * NHID + col] = hn;
    }
  }
}

// ---------- small helpers ----------

__global__ void f32_to_bf16_k(const float* __restrict__ src, bf16* __restrict__ dst, size_t n) {
  size_t i = (size_t)blockIdx.x * blockDim.x + threadIdx.x;
  size_t stride = (size_t)gridDim.x * blockDim.x;
  for (; i < n; i += stride) dst[i] = (bf16)src[i];
}

__global__ void alpha_init_k(const float* __restrict__ Wa, float* __restrict__ alpha) {
  int i = blockIdx.x * blockDim.x + threadIdx.x;
  if (i < NHID) alpha[i] = 1.f / (1.f + __expf(-Wa[i]));
}

__global__ void h_init_k(const float* __restrict__ hidden, float* __restrict__ hf,
                         bf16* __restrict__ hb) {
  int i = blockIdx.x * blockDim.x + threadIdx.x;
  if (i < B_ * NHID) { float v = hidden[i]; hf[i] = v; hb[i] = (bf16)v; }
}

// ---------- launch ----------

extern "C" void kernel_launch(void* const* d_in, const int* in_sizes, int n_in,
                              void* d_out, int out_size, void* d_ws, size_t ws_size,
                              hipStream_t stream) {
  (void)in_sizes; (void)n_in; (void)out_size; (void)ws_size;

  const float* input  = (const float*)d_in[0];
  const float* hidden = (const float*)d_in[1];
  const float* W_in   = (const float*)d_in[2];
  const float* b_in   = (const float*)d_in[3];
  const float* W_ih   = (const float*)d_in[4];
  const float* b_ih   = (const float*)d_in[5];
  const float* W_hh   = (const float*)d_in[6];
  const float* b_hh   = (const float*)d_in[7];
  const float* W_out  = (const float*)d_in[8];
  const float* b_out  = (const float*)d_in[9];
  const float* W_a    = (const float*)d_in[10];

  char* ws = (char*)d_ws;
  size_t off = 0;
  auto take = [&](size_t bytes) -> char* {
    char* p = ws + off;
    off += (bytes + 255) & ~(size_t)255;
    return p;
  };

  const size_t MT = (size_t)B_ * T_;
  bf16*  Win_bf   = (bf16*)take((size_t)NHID * NIN  * 2);
  bf16*  Wih_bf   = (bf16*)take((size_t)NHID * NHID * 2);
  bf16*  Whh_bf   = (bf16*)take((size_t)NHID * NHID * 2);
  bf16*  Wout_bf  = (bf16*)take((size_t)NOUT * NHID * 2);
  bf16*  Xin_bf   = (bf16*)take(MT * NIN  * 2);
  bf16*  X_bf     = (bf16*)take(MT * NHID * 2);
  float* Pre_f    = (float*)take(MT * NHID * 4);
  bf16*  Hlist_bf = (bf16*)take(MT * NHID * 2);
  float* Alpha_f  = (float*)take((size_t)NHID * 4);
  float* Hf0      = (float*)take((size_t)B_ * NHID * 4);
  float* Hf1      = (float*)take((size_t)B_ * NHID * 4);
  bf16*  Hb0      = (bf16*)take((size_t)B_ * NHID * 2);
  bf16*  Hb1      = (bf16*)take((size_t)B_ * NHID * 2);
  float* Hf[2] = {Hf0, Hf1};
  bf16*  Hb[2] = {Hb0, Hb1};

  float* out_hidden = (float*)d_out;                        // [B,T,H]
  float* out_output = out_hidden + MT * NHID;               // [B,T,O]
  float* out_hlast  = out_output + MT * NOUT;               // [B,H]

  auto cvt = [&](const float* s, bf16* d, size_t n) {
    int grid = (int)((n + 255) / 256);
    if (grid > 4096) grid = 4096;
    f32_to_bf16_k<<<grid, 256, 0, stream>>>(s, d, n);
  };
  cvt(W_in,  Win_bf,  (size_t)NHID * NIN);
  cvt(W_ih,  Wih_bf,  (size_t)NHID * NHID);
  cvt(W_hh,  Whh_bf,  (size_t)NHID * NHID);
  cvt(W_out, Wout_bf, (size_t)NOUT * NHID);
  cvt(input, Xin_bf,  MT * NIN);

  alpha_init_k<<<NHID / 256, 256, 0, stream>>>(W_a, Alpha_f);
  h_init_k<<<(B_ * NHID) / 256, 256, 0, stream>>>(hidden, Hf[0], Hb[0]);

  // x = input @ W_in^T + b_in   (bf16 out, feeds next GEMM)
  gemm_wmma_bf16<<<dim3(NHID / 64, MT / 64), 128, 0, stream>>>(
      Xin_bf, Win_bf, b_in, nullptr, X_bf, (int)MT, NHID, NIN);

  // pre = x @ W_ih^T + b_ih     (fp32 out, consumed by the scan)
  gemm_wmma_bf16<<<dim3(NHID / 64, MT / 64), 128, 0, stream>>>(
      X_bf, Wih_bf, b_ih, Pre_f, nullptr, (int)MT, NHID, NHID);

  // sequential scan: W_hh (2MB bf16) stays resident in L2 across steps
  for (int t = 0; t < T_; ++t) {
    rnn_step<<<NHID / 64, 128, 0, stream>>>(
        Hb[t & 1], Hf[t & 1], Whh_bf, Pre_f, b_hh, Alpha_f,
        Hb[(t + 1) & 1], Hf[(t + 1) & 1],
        out_hidden, Hlist_bf, out_hlast, t, (t == T_ - 1) ? 1 : 0);
  }

  // output_list = hidden_list @ W_out^T + b_out
  gemm_wmma_bf16<<<dim3(NOUT / 64, MT / 64), 128, 0, stream>>>(
      Hlist_bf, Wout_bf, b_out, out_output, nullptr, (int)MT, NOUT, NHID);
}
